// RVRT_15796889715239
// MI455X (gfx1250) — compile-verified, matrix operand order checked by reading, not run on hardware
//
#include <hip/hip_runtime.h>
#include <hip/hip_bf16.h>
#include <cstdint>

// ---------------------------------------------------------------------------
// RVRT / Swin-3D shifted-window attention block for gfx1250 (MI455X, wave32).
// GEMMs: V_WMMA_F32_16X16X32_F16 (f16 in, f32 acc), A-panels staged to LDS by
// the Tensor Data Mover (TENSOR_LOAD_TO_LDS, double-buffered, TENSORcnt sync).
// ---------------------------------------------------------------------------

typedef __attribute__((ext_vector_type(16))) _Float16 v16h;
typedef __attribute__((ext_vector_type(8)))  float    v8f;
typedef unsigned int u32x4 __attribute__((ext_vector_type(4)));
typedef int          i32x8 __attribute__((ext_vector_type(8)));
typedef int          i32x4 __attribute__((ext_vector_type(4)));

union AF { v16h v; uint4 q[2]; };

#define WMMA_F16(a, b, c) \
  __builtin_amdgcn_wmma_f32_16x16x32_f16(false, (a), false, (b), (short)0, (c), false, false)

#if __has_builtin(__builtin_amdgcn_tensor_load_to_lds) && \
    __has_builtin(__builtin_amdgcn_s_wait_tensorcnt)
#define HAVE_TDM 1
#else
#define HAVE_TDM 0
#endif

static __device__ __forceinline__ v8f v8f_zero() {
  v8f z;
  #pragma unroll
  for (int i = 0; i < 8; ++i) z[i] = 0.0f;
  return z;
}

// Problem constants
#define CH 192
#define NTOK 128
#define NWIN 512
#define NHEAD 6
#define HDIM 32
#define MROWS 65536
#define QSCALE 0.17677669529663687f   // 1/sqrt(32)
#define AP 72                          // padded LDS pitch in halves (144B)

#if HAVE_TDM
// One TDM op: DMA a 16-row x 64-col f16 tile (row stride K elems) from global
// into LDS with pad (32 DWORDs -> +4 DWORDs) so the LDS pitch becomes 72 halves.
// amdgpu-toolchain (clang-23) 6-arg form: (g0, g1, g2, g3, g_extra, cpol).
static __device__ __forceinline__
void tdm_load_a_slice(const _Float16* gsrc, unsigned ldsByteOff, int K) {
  const unsigned long long ga = (unsigned long long)(size_t)gsrc;
  u32x4 g0;
  g0[0] = 1u;                                                  // count=1 (valid)
  g0[1] = ldsByteOff;                                          // lds_addr
  g0[2] = (unsigned)ga;                                        // global_addr lo
  g0[3] = ((unsigned)(ga >> 32) & 0x01FFFFFFu) | (2u << 30);   // addr hi | type=2
  i32x8 g1;
  g1[0] = (int)((1u << 16) |            // data_size = 1 (2 bytes)
                (1u << 20) |            // pad_enable
                (4u << 22) |            // pad_interval: 32 DWORDs (one 128B row)
                (3u << 25));            // pad_amount: 4 DWORDs (16B)
  g1[1] = (int)(((unsigned)K & 0xFFFFu) << 16);   // tensor_dim0 = K
  g1[2] = 0;                                      // dim0 hi=0, tensor_dim1 lo=0
  g1[3] = (int)(1u | (64u << 16));                // tensor_dim1 hi (65536), tile_dim0=64
  g1[4] = 16;                                     // tile_dim1 = 16 rows / wave
  g1[5] = K;                                      // tensor_dim0_stride = K
  g1[6] = 0;
  g1[7] = 0;
  i32x4 z4; z4[0] = 0; z4[1] = 0; z4[2] = 0; z4[3] = 0;
  i32x8 z8;
  #pragma unroll
  for (int i = 0; i < 8; ++i) z8[i] = 0;
  __builtin_amdgcn_tensor_load_to_lds(g0, g1, z4, z4, z8, 0);
}
#endif

// ---------------------------------------------------------------------------
// Kernel 1: LN1 + cyclic shift (-1,-4,-4) + window partition, f32 -> f16.
// ---------------------------------------------------------------------------
__global__ __launch_bounds__(256)
void ln_shift_part_kernel(const float* __restrict__ x,
                          const float* __restrict__ lw,
                          const float* __restrict__ lb,
                          _Float16* __restrict__ xw) {
  const int tid  = threadIdx.x;
  const int wave = tid >> 5;
  const int lane = tid & 31;
  const long row = (long)blockIdx.x * 8 + wave;

  const long wIdx = row >> 7;  const int tok = (int)(row & 127);
  const int b  = (int)(wIdx >> 8);
  const int wp = (int)(wIdx & 255);
  const int dWn = wp >> 6, hWn = (wp >> 3) & 7, wWn = wp & 7;
  const int d1 = tok >> 6,  h1 = (tok >> 3) & 7, w1 = tok & 7;
  const int d = dWn * 2 + d1, h = hWn * 8 + h1, w = wWn * 8 + w1;
  const int ds = (d + 1) & 7, hs = (h + 4) & 63, wsrc = (w + 4) & 63;
  const float* src = x + (((long)(b * 8 + ds) * 64 + hs) * 64 + wsrc) * CH;

  float v[6];
  float s = 0.0f;
  #pragma unroll
  for (int i = 0; i < 6; ++i) { v[i] = src[lane + 32 * i]; s += v[i]; }
  #pragma unroll
  for (int off = 16; off >= 1; off >>= 1) s += __shfl_xor(s, off, 32);
  const float mean = s * (1.0f / CH);
  float vs = 0.0f;
  #pragma unroll
  for (int i = 0; i < 6; ++i) { const float dv = v[i] - mean; vs += dv * dv; }
  #pragma unroll
  for (int off = 16; off >= 1; off >>= 1) vs += __shfl_xor(vs, off, 32);
  const float inv = rsqrtf(vs * (1.0f / CH) + 1e-5f);

  _Float16* dst = xw + row * CH;
  #pragma unroll
  for (int i = 0; i < 6; ++i) {
    const int c = lane + 32 * i;
    dst[c] = (_Float16)((v[i] - mean) * inv * lw[c] + lb[c]);
  }
}

// ---------------------------------------------------------------------------
// Kernel 2 (LN2): plain LayerNorm on x2 (natural order), f32 -> f16.
// ---------------------------------------------------------------------------
__global__ __launch_bounds__(256)
void ln2_kernel(const float* __restrict__ x2,
                const float* __restrict__ lw,
                const float* __restrict__ lb,
                _Float16* __restrict__ h2) {
  const int tid  = threadIdx.x;
  const int wave = tid >> 5;
  const int lane = tid & 31;
  const long row = (long)blockIdx.x * 8 + wave;
  const float* src = x2 + row * CH;

  float v[6];
  float s = 0.0f;
  #pragma unroll
  for (int i = 0; i < 6; ++i) { v[i] = src[lane + 32 * i]; s += v[i]; }
  #pragma unroll
  for (int off = 16; off >= 1; off >>= 1) s += __shfl_xor(s, off, 32);
  const float mean = s * (1.0f / CH);
  float vs = 0.0f;
  #pragma unroll
  for (int i = 0; i < 6; ++i) { const float dv = v[i] - mean; vs += dv * dv; }
  #pragma unroll
  for (int off = 16; off >= 1; off >>= 1) vs += __shfl_xor(vs, off, 32);
  const float inv = rsqrtf(vs * (1.0f / CH) + 1e-5f);

  _Float16* dst = h2 + row * CH;
  #pragma unroll
  for (int i = 0; i < 6; ++i) {
    const int c = lane + 32 * i;
    dst[c] = (_Float16)((v[i] - mean) * inv * lw[c] + lb[c]);
  }
}

// ---------------------------------------------------------------------------
// Generic WMMA GEMM: A[M x K] f16 row-major (TDM-staged into LDS, pitch 72),
// Bw[K x N] f32 (converted to f16 while staging, stored Bt in LDS, pitch 72).
// Block tile 128x64, 8 waves as 4(M) x 2(N), 2x2 WMMA tiles per wave,
// K chunked by 64 (2 k-steps of 32), double-buffered TDM pipeline.
// Epilogues:
//   EPI 0: QKV split (+bias, q scaled) -> q/k/v [win][head][128][32] f16
//   EPI 1: proj (+bias) + window-reverse + unshift + residual -> x2 f32
//   EPI 2: fc1 (+bias) + exact GELU -> h3 f16
//   EPI 3: fc2 (+bias) + residual(x2) -> out f32
// ---------------------------------------------------------------------------
template <int EPI>
__global__ __launch_bounds__(256)
void gemm_wmma_kernel(const _Float16* __restrict__ A,
                      const float* __restrict__ Bw,
                      const float* __restrict__ bias,
                      int K, int N,
                      _Float16* __restrict__ o16a,
                      _Float16* __restrict__ o16b,
                      _Float16* __restrict__ o16c,
                      const float* __restrict__ fin,
                      float* __restrict__ fout) {
  __shared__ _Float16 As[2][128 * AP];   // 36.0 KB
  __shared__ _Float16 Bs[2][64 * AP];    // 18.0 KB

  const int tid  = threadIdx.x;
  const int wave = tid >> 5;
  const int lane = tid & 31;
  const int half = lane >> 4;
  const int l16  = lane & 15;
  const int wm   = wave >> 1;
  const int wn   = wave & 1;
  const long rowBase = (long)blockIdx.y * 128;
  const int  colBase = blockIdx.x * 64;
  const int  NC = K >> 6;
#if HAVE_TDM
  const unsigned wv = (unsigned)__builtin_amdgcn_readfirstlane(wave);
#endif

  v8f acc[2][2];
  #pragma unroll
  for (int i = 0; i < 2; ++i)
    #pragma unroll
    for (int j = 0; j < 2; ++j) acc[i][j] = v8f_zero();

  auto stageA = [&](int buf, int kb) {
#if HAVE_TDM
    // each of the 8 waves DMAs its own 16-row slice (wave-uniform descriptor)
    const unsigned ldsOff =
        (unsigned)(size_t)(void*)&As[buf][0] + wv * (16u * AP * 2u);
    tdm_load_a_slice(A + (rowBase + wv * 16) * (long)K + kb, ldsOff, K);
#else
    #pragma unroll
    for (int i = 0; i < 8; ++i) {
      const int u  = tid + 256 * i;       // uint2 (4-half) units, 2048 total
      const int r  = u >> 4;
      const int c4 = u & 15;
      *(uint2*)&As[buf][r * AP + c4 * 4] =
          *(const uint2*)&A[(rowBase + r) * (long)K + kb + c4 * 4];
    }
#endif
  };
  auto stageB = [&](int buf, int kb) {
    #pragma unroll
    for (int i = 0; i < 16; ++i) {
      const int f  = tid + 256 * i;       // 64x64 elems
      const int n  = f & 63;
      const int kk = f >> 6;
      Bs[buf][n * AP + kk] = (_Float16)Bw[(long)(kb + kk) * N + colBase + n];
    }
  };

  // prologue: chunk 0
  stageA(0, 0);
  stageB(0, 0);
#if HAVE_TDM
  __builtin_amdgcn_s_wait_tensorcnt(0);
#endif
  __syncthreads();

  for (int c = 0; c < NC; ++c) {
    const int cur = c & 1;
    const int nxt = cur ^ 1;
    if (c + 1 < NC) {
      stageA(nxt, (c + 1) << 6);          // async TDM overlaps compute below
      if (c + 2 < NC)
        __builtin_prefetch(&Bw[(long)((c + 2) << 6) * N + colBase], 0, 0);
      stageB(nxt, (c + 1) << 6);
    }

    #pragma unroll
    for (int ks = 0; ks < 2; ++ks) {
      AF a[2], b[2];
      #pragma unroll
      for (int i = 0; i < 2; ++i) {
        const int m = wm * 32 + i * 16 + l16;
        a[i].q[0] = *(const uint4*)&As[cur][m * AP + ks * 32 + half * 8];
        a[i].q[1] = *(const uint4*)&As[cur][m * AP + ks * 32 + 16 + half * 8];
      }
      #pragma unroll
      for (int j = 0; j < 2; ++j) {
        const int n = wn * 32 + j * 16 + l16;
        b[j].q[0] = *(const uint4*)&Bs[cur][n * AP + ks * 32 + half * 8];
        b[j].q[1] = *(const uint4*)&Bs[cur][n * AP + ks * 32 + 16 + half * 8];
      }
      #pragma unroll
      for (int i = 0; i < 2; ++i)
        #pragma unroll
        for (int j = 0; j < 2; ++j)
          acc[i][j] = WMMA_F16(a[i].v, b[j].v, acc[i][j]);
    }

#if HAVE_TDM
    __builtin_amdgcn_s_wait_tensorcnt(0); // next A slice landed (per wave)
#endif
    __syncthreads();                      // publish As[nxt]/Bs[nxt]
  }

  // ---------------- epilogue ----------------
  #pragma unroll
  for (int i = 0; i < 2; ++i) {
    #pragma unroll
    for (int j = 0; j < 2; ++j) {
      #pragma unroll
      for (int r = 0; r < 8; ++r) {
        const long grow = rowBase + wm * 32 + i * 16 + r + 8 * half;
        const int  gcol = colBase + wn * 32 + j * 16 + l16;
        float val = acc[i][j][r] + bias[gcol];

        if constexpr (EPI == 0) {          // QKV split
          const int which = gcol / CH;
          const int hc    = gcol % CH;
          const int head  = hc >> 5;
          const int dd    = hc & 31;
          const long wIdx = grow >> 7;
          const int  tok  = (int)(grow & 127);
          const long dst  = (((wIdx * NHEAD + head) * NTOK + tok) << 5) + dd;
          if (which == 0)      o16a[dst] = (_Float16)(val * QSCALE);
          else if (which == 1) o16b[dst] = (_Float16)val;
          else                 o16c[dst] = (_Float16)val;
        } else if constexpr (EPI == 1) {   // proj + reverse + unshift + residual
          const long wIdx = grow >> 7;
          const int  tok  = (int)(grow & 127);
          const int  b    = (int)(wIdx >> 8);
          const int  wp   = (int)(wIdx & 255);
          const int dWn = wp >> 6, hWn = (wp >> 3) & 7, wWn = wp & 7;
          const int d1 = tok >> 6, h1 = (tok >> 3) & 7, w1 = tok & 7;
          const int d = dWn * 2 + d1, h = hWn * 8 + h1, w = wWn * 8 + w1;
          const int dd_ = (d + 1) & 7, hd_ = (h + 4) & 63, wd_ = (w + 4) & 63;
          const long idx =
              (((long)(b * 8 + dd_) * 64 + hd_) * 64 + wd_) * CH + gcol;
          fout[idx] = fin[idx] + val;
        } else if constexpr (EPI == 2) {   // fc1 + exact GELU
          const float g = 0.5f * val * (1.0f + erff(val * 0.70710678f));
          o16a[grow * (long)N + gcol] = (_Float16)g;
        } else {                           // fc2 + residual -> final out
          const long idx = grow * (long)N + gcol;
          fout[idx] = fin[idx] + val;
        }
      }
    }
  }
}

// ---------------------------------------------------------------------------
// Kernel 3: per-(window, head) attention.
//  S = q.k^T (8 WMMAs/wave) + analytic rel-pos bias + analytic shift mask,
//  f32 softmax (16-lane shfl groups), O = P.V (16 WMMAs/wave).
// ---------------------------------------------------------------------------
__global__ __launch_bounds__(256)
void attn_kernel(const _Float16* __restrict__ q,
                 const _Float16* __restrict__ k,
                 const _Float16* __restrict__ v,
                 const float* __restrict__ rpb,
                 _Float16* __restrict__ out) {
  __shared__ _Float16 qs[128 * 32];
  __shared__ _Float16 kt[128 * 32];
  __shared__ _Float16 vt[32 * 136];
  __shared__ _Float16 Ps[128 * 136];
  __shared__ float    rps[675];
  __shared__ unsigned char lbl[128];

  const int tid  = threadIdx.x;
  const int wave = tid >> 5;
  const int lane = tid & 31;
  const int half = lane >> 4;
  const int l16  = lane & 15;

  const int  blk  = blockIdx.x;
  const long wIdx = blk / NHEAD;
  const int  head = blk % NHEAD;
  const _Float16* qg = q + (long)blk * (NTOK * HDIM);
  const _Float16* kg = k + (long)blk * (NTOK * HDIM);
  const _Float16* vg = v + (long)blk * (NTOK * HDIM);

  {
    const int o = tid * 16;
    *(uint4*)&qs[o]     = *(const uint4*)&qg[o];
    *(uint4*)&qs[o + 8] = *(const uint4*)&qg[o + 8];
    *(uint4*)&kt[o]     = *(const uint4*)&kg[o];
    *(uint4*)&kt[o + 8] = *(const uint4*)&kg[o + 8];
  }
  #pragma unroll
  for (int i = 0; i < 16; ++i) {
    const int f   = tid + 256 * i;
    const int tok = f >> 5;
    const int dd  = f & 31;
    vt[dd * 136 + tok] = vg[f];
  }
  #pragma unroll
  for (int i = 0; i < 3; ++i) {
    const int f = tid + 256 * i;
    if (f < 675) rps[f] = rpb[f * NHEAD + head];
  }
  if (tid < 128) {
    const int wp  = (int)(wIdx & 255);
    const int dWn = wp >> 6, hWn = (wp >> 3) & 7, wWn = wp & 7;
    const int d1 = tid >> 6, h1 = (tid >> 3) & 7, w1 = tid & 7;
    const int gd = dWn * 2 + d1, gh = hWn * 8 + h1, gw = wWn * 8 + w1;
    const int Ld = gd < 6 ? 0 : (gd < 7 ? 1 : 2);
    const int Lh = gh < 56 ? 0 : (gh < 60 ? 1 : 2);
    const int Lw = gw < 56 ? 0 : (gw < 60 ? 1 : 2);
    lbl[tid] = (unsigned char)(Ld * 9 + Lh * 3 + Lw);
  }
  __syncthreads();

  AF aq;
  {
    const int m = wave * 16 + l16;
    aq.q[0] = *(const uint4*)&qs[m * 32 + half * 8];
    aq.q[1] = *(const uint4*)&qs[m * 32 + 16 + half * 8];
  }
  v8f s[8];
  #pragma unroll
  for (int nT = 0; nT < 8; ++nT) {
    AF bk;
    const int n = nT * 16 + l16;
    bk.q[0] = *(const uint4*)&kt[n * 32 + half * 8];
    bk.q[1] = *(const uint4*)&kt[n * 32 + 16 + half * 8];
    s[nT] = WMMA_F16(aq.v, bk.v, v8f_zero());
  }

  #pragma unroll
  for (int nT = 0; nT < 8; ++nT) {
    const int jj = nT * 16 + l16;
    const int d2 = jj >> 6, h2c = (jj >> 3) & 7, w2 = jj & 7;
    const unsigned char lj = lbl[jj];
    #pragma unroll
    for (int r = 0; r < 8; ++r) {
      const int ii = wave * 16 + r + 8 * half;
      const int d1 = ii >> 6, h1 = (ii >> 3) & 7, w1 = ii & 7;
      const int idx = (d1 - d2 + 1) * 225 + (h1 - h2c + 7) * 15 + (w1 - w2 + 7);
      float add = rps[idx];
      if (lbl[ii] != lj) add -= 100.0f;
      s[nT][r] += add;
    }
  }

  float mx[8], sm[8];
  #pragma unroll
  for (int r = 0; r < 8; ++r) {
    float m = s[0][r];
    #pragma unroll
    for (int nT = 1; nT < 8; ++nT) m = fmaxf(m, s[nT][r]);
    #pragma unroll
    for (int off = 8; off >= 1; off >>= 1) m = fmaxf(m, __shfl_xor(m, off, 16));
    mx[r] = m;
  }
  #pragma unroll
  for (int nT = 0; nT < 8; ++nT)
    #pragma unroll
    for (int r = 0; r < 8; ++r) s[nT][r] = __expf(s[nT][r] - mx[r]);
  #pragma unroll
  for (int r = 0; r < 8; ++r) {
    float t = s[0][r];
    #pragma unroll
    for (int nT = 1; nT < 8; ++nT) t += s[nT][r];
    #pragma unroll
    for (int off = 8; off >= 1; off >>= 1) t += __shfl_xor(t, off, 16);
    sm[r] = 1.0f / t;
  }
  #pragma unroll
  for (int nT = 0; nT < 8; ++nT) {
    #pragma unroll
    for (int r = 0; r < 8; ++r) {
      const int ii = wave * 16 + r + 8 * half;
      Ps[ii * 136 + nT * 16 + l16] = (_Float16)(s[nT][r] * sm[r]);
    }
  }
  __syncthreads();

  v8f o[2];
  o[0] = v8f_zero();
  o[1] = v8f_zero();
  #pragma unroll
  for (int kb = 0; kb < 8; ++kb) {
    AF ap;
    const int mm = wave * 16 + l16;
    ap.q[0] = *(const uint4*)&Ps[mm * 136 + kb * 32 + half * 8];
    ap.q[1] = *(const uint4*)&Ps[mm * 136 + kb * 32 + 16 + half * 8];
    #pragma unroll
    for (int j = 0; j < 2; ++j) {
      AF bv;
      const int n = j * 16 + l16;
      bv.q[0] = *(const uint4*)&vt[n * 136 + kb * 32 + half * 8];
      bv.q[1] = *(const uint4*)&vt[n * 136 + kb * 32 + 16 + half * 8];
      o[j] = WMMA_F16(ap.v, bv.v, o[j]);
    }
  }
  #pragma unroll
  for (int j = 0; j < 2; ++j) {
    #pragma unroll
    for (int r = 0; r < 8; ++r) {
      const int ii = wave * 16 + r + 8 * half;
      out[(wIdx * NTOK + ii) * (long)CH + head * HDIM + j * 16 + l16] =
          (_Float16)o[j][r];
    }
  }
}

// ---------------------------------------------------------------------------
// Host launcher
// ---------------------------------------------------------------------------
extern "C" void kernel_launch(void* const* d_in, const int* in_sizes, int n_in,
                              void* d_out, int out_size, void* d_ws, size_t ws_size,
                              hipStream_t stream) {
  (void)in_sizes; (void)n_in; (void)out_size; (void)ws_size;

  const float* x      = (const float*)d_in[0];
  const float* qkv_w  = (const float*)d_in[1];
  const float* qkv_b  = (const float*)d_in[2];
  const float* proj_w = (const float*)d_in[3];
  const float* proj_b = (const float*)d_in[4];
  const float* rpb    = (const float*)d_in[5];
  const float* ln1_w  = (const float*)d_in[6];
  const float* ln1_b  = (const float*)d_in[7];
  const float* ln2_w  = (const float*)d_in[8];
  const float* ln2_b  = (const float*)d_in[9];
  const float* fc1_w  = (const float*)d_in[10];
  const float* fc1_b  = (const float*)d_in[11];
  const float* fc2_w  = (const float*)d_in[12];
  const float* fc2_b  = (const float*)d_in[13];
  float* out = (float*)d_out;

  char* ws = (char*)d_ws;
  const size_t SZ16 = (size_t)MROWS * CH * 2;
  _Float16* xw  = (_Float16*)(ws);
  _Float16* qb  = (_Float16*)(ws + 1 * SZ16);
  _Float16* kb  = (_Float16*)(ws + 2 * SZ16);
  _Float16* vb  = (_Float16*)(ws + 3 * SZ16);
  float*    x2  = (float*)   (ws + 4 * SZ16);
  _Float16* ao  = xw;   // attn out aliases xw
  _Float16* h2  = qb;   // LN2 out aliases q
  _Float16* h3  = kb;   // fc1 out (384-wide) aliases k..v

  ln_shift_part_kernel<<<MROWS / 8, 256, 0, stream>>>(x, ln1_w, ln1_b, xw);

  gemm_wmma_kernel<0><<<dim3(576 / 64, MROWS / 128), 256, 0, stream>>>(
      xw, qkv_w, qkv_b, 192, 576, qb, kb, vb, nullptr, nullptr);

  attn_kernel<<<NWIN * NHEAD, 256, 0, stream>>>(qb, kb, vb, rpb, ao);

  gemm_wmma_kernel<1><<<dim3(192 / 64, MROWS / 128), 256, 0, stream>>>(
      ao, proj_w, proj_b, 192, 192, nullptr, nullptr, nullptr, x, x2);

  ln2_kernel<<<MROWS / 8, 256, 0, stream>>>(x2, ln2_w, ln2_b, h2);

  gemm_wmma_kernel<2><<<dim3(384 / 64, MROWS / 128), 256, 0, stream>>>(
      h2, fc1_w, fc1_b, 192, 384, h3, nullptr, nullptr, nullptr, nullptr);

  gemm_wmma_kernel<3><<<dim3(192 / 64, MROWS / 128), 256, 0, stream>>>(
      h3, fc2_w, fc2_b, 384, 192, nullptr, nullptr, nullptr, x2, out);
}
